// SimpleLSTM_57397942944421
// MI455X (gfx1250) — compile-verified, matrix operand order checked by reading, not run on hardware
//
#include <hip/hip_runtime.h>
#include <math.h>

#define FEAT   40
#define HID    128
#define GATES  512          // 4*HID
#define BATCH  128
#define TLEN   2048
#define BT     (BATCH * TLEN)

typedef __attribute__((ext_vector_type(16))) _Float16 v16h;
typedef __attribute__((ext_vector_type(8)))  _Float16 v8h;
typedef __attribute__((ext_vector_type(4)))  _Float16 v4h;
typedef __attribute__((ext_vector_type(8)))  float    v8f;

union V16 { v16h v; v8h h[2]; };

// ---------------------------------------------------------------------------
// f32 -> f16 convert with optional zero-padding of the K dimension.
// Launch: <<<Nrows, Kout>>>
// ---------------------------------------------------------------------------
__global__ void k_cvt_pad(const float* __restrict__ src, _Float16* __restrict__ dst,
                          int Kin, int Kout) {
  const int row = blockIdx.x;
  const int k   = threadIdx.x;
  float v = (k < Kin) ? src[(size_t)row * Kin + k] : 0.0f;
  dst[(size_t)row * Kout + k] = (_Float16)v;
}

// ---------------------------------------------------------------------------
// Pre-GEMM: gx[M, 512] = A[M, K] @ W[512, K]^T   (f16 in, f16 out, f32 acc)
// Block: 256 threads (8 waves). Each block handles one 16-row M tile; each
// wave covers 64 output columns (4 WMMA N-tiles).
// ---------------------------------------------------------------------------
__global__ __launch_bounds__(256)
void k_pregemm(const _Float16* __restrict__ A,
               const _Float16* __restrict__ W,
               _Float16* __restrict__ gx,
               int K) {
  const int lane  = threadIdx.x & 31;
  const int wave  = threadIdx.x >> 5;
  const int mBase = blockIdx.x * 16;
  const int nBase = wave * 64;
  const int mLane = lane & 15;
  const int ksel  = (lane >> 4) * 8;      // 16-bit A/B fragment half-select

  v8f acc[4] = {};
  const int kTiles = K >> 5;
  for (int kt = 0; kt < kTiles; ++kt) {
    const int k0 = kt << 5;
    V16 a;
    const _Float16* ap = A + (size_t)(mBase + mLane) * K + k0 + ksel;
    a.h[0] = *(const v8h*)(ap);
    a.h[1] = *(const v8h*)(ap + 16);
#pragma unroll
    for (int j = 0; j < 4; ++j) {
      const int n = nBase + j * 16 + mLane;
      V16 b;
      const _Float16* bp = W + (size_t)n * K + k0 + ksel;
      b.h[0] = *(const v8h*)(bp);
      b.h[1] = *(const v8h*)(bp + 16);
      acc[j] = __builtin_amdgcn_wmma_f32_16x16x32_f16(
          false, a.v, false, b.v, (short)0, acc[j], false, false);
    }
  }
  // D layout: VGPR r, lanes 0-15 -> M=r, lanes 16-31 -> M=8+r; N = lane&15
#pragma unroll
  for (int j = 0; j < 4; ++j) {
    const int n = nBase + j * 16 + mLane;
#pragma unroll
    for (int r = 0; r < 8; ++r) {
      const int m = mBase + r + ((lane >> 4) << 3);
      gx[(size_t)m * GATES + n] = (_Float16)acc[j][r];
    }
  }
}

// ---------------------------------------------------------------------------
// Recurrent LSTM layer. Grid: 8 blocks (16 batch rows each), 512 threads
// (16 waves; each wave owns 2 WMMA N-tiles = 32 gate columns).
// w_hh fragments are weight-stationary in VGPRs across all 2048 steps.
// ---------------------------------------------------------------------------
__global__ __launch_bounds__(512)
void k_recurrent(const _Float16* __restrict__ gx,    // [B*T, 512]
                 const _Float16* __restrict__ Whh,   // [512, 128] row-major
                 const float* __restrict__ bih,      // [512]
                 const float* __restrict__ bhh,      // [512]
                 _Float16* __restrict__ hout,        // [B, T, 128]
                 float* __restrict__ hT) {           // [B, 128]
  __shared__ __align__(16) _Float16 h_lds[16][HID];
  __shared__ __align__(16) float    gates[16][GATES];
  __shared__ float bias[GATES];

  const int tid   = threadIdx.x;
  const int lane  = tid & 31;
  const int wave  = tid >> 5;
  const int bTile = blockIdx.x;
  const int nBase = wave * 32;
  const int mLane = lane & 15;
  const int ksel  = (lane >> 4) * 8;

  for (int i = tid; i < 16 * HID; i += 512) ((_Float16*)h_lds)[i] = (_Float16)0.0f;
  for (int i = tid; i < GATES;    i += 512) bias[i] = bih[i] + bhh[i];

  // Hoist w_hh B-fragments into registers: [kTile][nTile] = 8 frags = 64 VGPRs.
  V16 bfr[4][2];
#pragma unroll
  for (int kt = 0; kt < 4; ++kt) {
#pragma unroll
    for (int j = 0; j < 2; ++j) {
      const int n  = nBase + j * 16 + mLane;
      const int k0 = kt << 5;
      const _Float16* bp = Whh + (size_t)n * HID + k0 + ksel;
      bfr[kt][j].h[0] = *(const v8h*)(bp);
      bfr[kt][j].h[1] = *(const v8h*)(bp + 16);
    }
  }

  // Per-thread persistent cell state: row = tid>>5, cols (tid&31)*4 .. +4
  const int row  = tid >> 5;
  const int colb = (lane) * 4;
  const int bRow = bTile * 16 + row;
  const _Float16* gxRow = gx + (size_t)bRow * TLEN * GATES;
  float c[4];
#pragma unroll
  for (int j = 0; j < 4; ++j) c[j] = 0.0f;

  __syncthreads();

  for (int t = 0; t < TLEN; ++t) {
    // Issue gx loads for this step up front: independent of the GEMM, so
    // they retire under the WMMA + barrier shadow.
    const _Float16* gxp = gxRow + (size_t)t * GATES;
    v4h gi = *(const v4h*)(gxp + colb);
    v4h gf = *(const v4h*)(gxp + HID     + colb);
    v4h gg = *(const v4h*)(gxp + 2 * HID + colb);
    v4h go = *(const v4h*)(gxp + 3 * HID + colb);
    if (t + 8 < TLEN)   // stream gx ahead into near caches (global_prefetch)
      __builtin_prefetch(gxRow + (size_t)(t + 8) * GATES + colb, 0, 1);

    // ---- phase A: gates_mm = h @ Whh^T (16 x 512 spread over 16 waves) ----
    v8f acc[2] = {};
#pragma unroll
    for (int kt = 0; kt < 4; ++kt) {
      const int k0 = kt << 5;
      V16 a;
      const _Float16* ap = &h_lds[mLane][k0 + ksel];
      a.h[0] = *(const v8h*)(ap);
      a.h[1] = *(const v8h*)(ap + 16);
#pragma unroll
      for (int j = 0; j < 2; ++j) {
        acc[j] = __builtin_amdgcn_wmma_f32_16x16x32_f16(
            false, a.v, false, bfr[kt][j].v, (short)0, acc[j], false, false);
      }
    }
#pragma unroll
    for (int j = 0; j < 2; ++j) {
      const int n = nBase + j * 16 + mLane;
#pragma unroll
      for (int r = 0; r < 8; ++r)
        gates[r + ((lane >> 4) << 3)][n] = acc[j][r];
    }
    __syncthreads();

    // ---- phase B: elementwise gate update (i,f,g,o) ----
    v4h hv4;
#pragma unroll
    for (int j = 0; j < 4; ++j) {
      const int n = colb + j;
      float iv = gates[row][n]           + (float)gi[j] + bias[n];
      float fv = gates[row][HID + n]     + (float)gf[j] + bias[HID + n];
      float gv = gates[row][2 * HID + n] + (float)gg[j] + bias[2 * HID + n];
      float ov = gates[row][3 * HID + n] + (float)go[j] + bias[3 * HID + n];
      float si = 1.0f / (1.0f + __expf(-iv));
      float sf = 1.0f / (1.0f + __expf(-fv));
      float so = 1.0f / (1.0f + __expf(-ov));
      float tg = tanhf(gv);
      c[j] = sf * c[j] + si * tg;
      float hv = so * tanhf(c[j]);
      hv4[j] = (_Float16)hv;
      if (t == TLEN - 1) hT[(size_t)bRow * HID + n] = hv;
    }
    *(v4h*)&h_lds[row][colb] = hv4;
    *(v4h*)(hout + ((size_t)bRow * TLEN + t) * HID + colb) = hv4;
    __syncthreads();
  }
}

// ---------------------------------------------------------------------------
// Final projection: out = hT @ w_out^T + b_out ; tuple (out, out) duplicated.
// ---------------------------------------------------------------------------
__global__ void k_final(const float* __restrict__ hT, const float* __restrict__ wout,
                        const float* __restrict__ bout, float* __restrict__ out) {
  const int tid = threadIdx.x;           // 256 = 128 batch * 2 outputs
  const int b = tid >> 1, o = tid & 1;
  float s = bout[o];
  const float* hp = hT + (size_t)b * HID;
  const float* wp = wout + (size_t)o * HID;
  for (int k = 0; k < HID; ++k) s += hp[k] * wp[k];
  out[b * 2 + o] = s;
  out[256 + b * 2 + o] = s;
}

// ---------------------------------------------------------------------------
extern "C" void kernel_launch(void* const* d_in, const int* in_sizes, int n_in,
                              void* d_out, int out_size, void* d_ws, size_t ws_size,
                              hipStream_t stream) {
  (void)in_sizes; (void)n_in; (void)out_size; (void)ws_size;
  const float* x         = (const float*)d_in[0];  // [128,1,2048,40]
  const float* w_ih0     = (const float*)d_in[1];  // [512,40]
  const float* w_ih_rest = (const float*)d_in[2];  // [3,512,128]
  const float* w_hh      = (const float*)d_in[3];  // [4,512,128]
  const float* b_ih      = (const float*)d_in[4];  // [4,512]
  const float* b_hh      = (const float*)d_in[5];  // [4,512]
  const float* w_out     = (const float*)d_in[6];  // [2,128]
  const float* b_out     = (const float*)d_in[7];  // [2]

  char* ws = (char*)d_ws;
  size_t off = 0;
  _Float16* xp  = (_Float16*)(ws + off); off += (size_t)BT * 64 * 2;     // padded x
  _Float16* hA  = (_Float16*)(ws + off); off += (size_t)BT * HID * 2;
  _Float16* hB  = (_Float16*)(ws + off); off += (size_t)BT * HID * 2;
  _Float16* gxb = (_Float16*)(ws + off); off += (size_t)BT * GATES * 2;
  _Float16* wIn = (_Float16*)(ws + off); off += (size_t)GATES * HID * 2;
  _Float16* wHH = (_Float16*)(ws + off); off += (size_t)GATES * HID * 2;
  float*    hT  = (float*)(ws + off);    off += (size_t)BATCH * HID * 4;

  // Layer 0 input: f32 [B*T, 40] -> f16 [B*T, 64] zero-padded
  k_cvt_pad<<<BT, 64, 0, stream>>>(x, xp, FEAT, 64);

  _Float16* hbufs[2] = {hA, hB};
  for (int l = 0; l < 4; ++l) {
    const int K = (l == 0) ? 64 : HID;
    const float* wihsrc = (l == 0) ? w_ih0
                                   : (w_ih_rest + (size_t)(l - 1) * GATES * HID);
    const int KinW = (l == 0) ? FEAT : HID;
    k_cvt_pad<<<GATES, K,   0, stream>>>(wihsrc, wIn, KinW, K);
    k_cvt_pad<<<GATES, HID, 0, stream>>>(w_hh + (size_t)l * GATES * HID, wHH, HID, HID);

    const _Float16* Ain = (l == 0) ? xp : hbufs[(l - 1) & 1];
    _Float16* Hout = hbufs[l & 1];

    k_pregemm<<<BT / 16, 256, 0, stream>>>(Ain, wIn, gxb, K);
    k_recurrent<<<8, 512, 0, stream>>>(gxb, wHH,
                                       b_ih + (size_t)l * GATES,
                                       b_hh + (size_t)l * GATES,
                                       Hout, hT);
  }
  k_final<<<1, 256, 0, stream>>>(hT, w_out, b_out, (float*)d_out);
}
